// Decoder_14663018348866
// MI455X (gfx1250) — compile-verified, matrix operand order checked by reading, not run on hardware
//
#include <hip/hip_runtime.h>

// ---------------------------------------------------------------------------
// Types
// ---------------------------------------------------------------------------
typedef __attribute__((ext_vector_type(16))) __bf16         v16bf;
typedef __attribute__((ext_vector_type(8)))  float          v8f;
typedef __attribute__((ext_vector_type(8)))  unsigned short u16x8;

#define NEGBIG (-1.0e9f)

__device__ __forceinline__ unsigned short f2bf(float x) {
  union { float f; unsigned int u; } c; c.f = x;
  unsigned int u = c.u;
  unsigned int r = u + 0x7FFFu + ((u >> 16) & 1u);
  if ((u & 0x7F800000u) == 0x7F800000u) r = u;  // inf/nan passthrough
  return (unsigned short)(r >> 16);
}

// A-fragment (16x32 bf16) or B-fragment-from-transposed-source:
// frag element (sel = lane&15, K) = src[sel*ld + Kmap], Kmap = {8h+e, 16+8h+e}
__device__ __forceinline__ v16bf load_frag_rm(const unsigned short* p, int ld, int lane) {
  int sel = lane & 15, h = (lane >> 4) & 1;
  const unsigned short* r = p + (size_t)sel * ld + 8 * h;
  union { struct { u16x8 lo, hi; } s; v16bf v; } f;
  f.s.lo = *(const u16x8*)(r);
  f.s.hi = *(const u16x8*)(r + 16);
  return f.v;
}

__device__ __forceinline__ v8f wmma_bf16(v16bf a, v16bf b, v8f c) {
  return __builtin_amdgcn_wmma_f32_16x16x32_bf16(false, a, false, b, (short)0, c,
                                                 false, false);
}

// ---------------------------------------------------------------------------
// gfx1250 async direct-to-LDS staging (ASYNCcnt-tracked)
// ---------------------------------------------------------------------------
__device__ __forceinline__ void async_ld_b128(unsigned lds_byte_addr,
                                              const unsigned short* gaddr) {
  asm volatile("global_load_async_to_lds_b128 %0, %1, off"
               :: "v"(lds_byte_addr), "v"(gaddr)
               : "memory");
}
__device__ __forceinline__ void wait_async0() {
  asm volatile("s_wait_asynccnt 0" ::: "memory");
}
// Release address VGPRs of in-flight async ops before the register allocator
// recycles them (XNACK replay may re-sample addresses; tracked by XCNT).
__device__ __forceinline__ void wait_xcnt0() {
  asm volatile("s_wait_xcnt 0" ::: "memory");
}

// Stage one 64-byte-per-lane strip (4 x b128) of a weight row into LDS.
__device__ __forceinline__ void stage_async4(const unsigned short* wrow,
                                             unsigned lds_byte_addr) {
#pragma unroll
  for (int i = 0; i < 4; ++i)
    async_ld_b128(lds_byte_addr + i * 16, wrow + i * 8);
  wait_xcnt0();
}

// ---------------------------------------------------------------------------
// Shared GEMM core: C[128x64] += A[128x1024]bf16 * W^T ; W row-major [n][k].
// 8 waves; wave computes 16(M) x 64(N). K-tile 128, double-buffered LDS,
// async-to-LDS weight staging. acc[4] per wave.
// ---------------------------------------------------------------------------
__device__ __forceinline__ void gemm_128x64_core(
    const unsigned short* __restrict__ ap,    // A + m0*1024
    const unsigned short* __restrict__ wn0,   // W + n0*1024
    unsigned short* lds,                      // [2][64*128] ushort (32KB)
    v8f* acc, int lane, int tid) {
  const int sn = tid >> 2;             // 0..63  (n row within tile)
  const int kq = (tid & 3) * 32;       // 0,32,64,96 (k element offset)
  const unsigned ldsbase = (unsigned)(uintptr_t)lds;
  const unsigned stoff = ldsbase + (unsigned)(sn * 128 + kq) * 2;
  const unsigned short* wrow = wn0 + (size_t)sn * 1024 + kq;

  stage_async4(wrow, stoff);           // tile 0 -> buffer 0
  for (int kt = 0; kt < 8; ++kt) {
    const int buf = kt & 1;
    wait_async0();
    __syncthreads();
    if (kt < 7)
      stage_async4(wrow + (kt + 1) * 128, stoff + (buf ^ 1) * 16384);
    const unsigned short* lbuf = lds + buf * 8192;
#pragma unroll
    for (int kk = 0; kk < 4; ++kk) {
      v16bf af = load_frag_rm(ap + kt * 128 + kk * 32, 1024, lane);
#pragma unroll
      for (int t = 0; t < 4; ++t) {
        v16bf bfrg = load_frag_rm(lbuf + (t * 16) * 128 + kk * 32, 128, lane);
        acc[t] = wmma_bf16(af, bfrg, acc[t]);
      }
    }
  }
}

// ---------------------------------------------------------------------------
// 0) float -> bf16 conversion (weights)
// ---------------------------------------------------------------------------
__global__ void cvt_bf16_kernel(const float* __restrict__ src,
                                unsigned short* __restrict__ dst, int n) {
  int i = blockIdx.x * blockDim.x + threadIdx.x;
  if (i < n) dst[i] = f2bf(src[i]);
}

// ---------------------------------------------------------------------------
// 1) zpart[b][o] = dense_b[o] + sum_i z[b][i] * dense_w[o][16+i]
// ---------------------------------------------------------------------------
__global__ __launch_bounds__(256) void zpart_kernel(
    const float* __restrict__ z, const float* __restrict__ dw,
    const float* __restrict__ db, float* __restrict__ zp) {
  int o = blockIdx.x * 256 + threadIdx.x;  // 0..1023
  int b = blockIdx.y;                      // 0..7
  const float* zr = z + b * 768;
  const float* wr = dw + (size_t)o * 784 + 16;
  float s = db[o];
  for (int i = 0; i < 768; ++i) s += zr[i] * wr[i];
  zp[b * 1024 + o] = s;
}

// ---------------------------------------------------------------------------
// 2) x = dense(concat) ; xn = LayerNorm(x)   (one token per block, 256 thr)
// ---------------------------------------------------------------------------
__global__ __launch_bounds__(256) void dense_ln_kernel(
    const float* __restrict__ rt, const float* __restrict__ dw,
    const float* __restrict__ zp, const float* __restrict__ g,
    const float* __restrict__ bb, float* __restrict__ x,
    unsigned short* __restrict__ xn) {
  int tok = blockIdx.x;  // 0..8191
  int b = tok >> 10;
  __shared__ float rsh[16];
  __shared__ float red[2];
  __shared__ float s8[8], ss8[8];
  if (threadIdx.x < 16) rsh[threadIdx.x] = rt[tok * 16 + threadIdx.x];
  __syncthreads();
  float v[4];
  float s = 0.f, ss = 0.f;
#pragma unroll
  for (int j = 0; j < 4; ++j) {
    int o = threadIdx.x * 4 + j;
    const float* wr = dw + (size_t)o * 784;
    float a = zp[b * 1024 + o];
#pragma unroll
    for (int i = 0; i < 16; ++i) a += rsh[i] * wr[i];
    v[j] = a; s += a; ss += a * a;
  }
#pragma unroll
  for (int off = 16; off; off >>= 1) {
    s += __shfl_xor(s, off, 32);
    ss += __shfl_xor(ss, off, 32);
  }
  int wv = threadIdx.x >> 5;
  if ((threadIdx.x & 31) == 0) { s8[wv] = s; ss8[wv] = ss; }
  __syncthreads();
  if (threadIdx.x == 0) {
    float S1 = 0.f, S2 = 0.f;
    for (int i = 0; i < 8; ++i) { S1 += s8[i]; S2 += ss8[i]; }
    float mu = S1 * (1.0f / 1024.0f);
    red[0] = mu;
    red[1] = S2 * (1.0f / 1024.0f) - mu * mu;
  }
  __syncthreads();
  float mu = red[0];
  float rstd = rsqrtf(red[1] + 1e-6f);
#pragma unroll
  for (int j = 0; j < 4; ++j) {
    int o = threadIdx.x * 4 + j;
    float xx = v[j];
    x[(size_t)tok * 1024 + o] = xx;
    xn[(size_t)tok * 1024 + o] = f2bf(g[o] * (xx - mu) * rstd + bb[o]);
  }
}

// ---------------------------------------------------------------------------
// 3) QKV projection GEMM -> q,k bf16 [B*H,S,64], v^T bf16 [B*H,64,S]
// ---------------------------------------------------------------------------
__global__ __launch_bounds__(256) void qkv_gemm_kernel(
    const unsigned short* __restrict__ xn, const unsigned short* __restrict__ wqkv,
    const float* __restrict__ bq, const float* __restrict__ bk,
    const float* __restrict__ bv, unsigned short* __restrict__ q_out,
    unsigned short* __restrict__ k_out, unsigned short* __restrict__ vt_out) {
  __shared__ unsigned short wtile[2][64 * 128];
  const int lane = threadIdx.x & 31, wave = threadIdx.x >> 5;
  const int m0 = blockIdx.x * 128 + wave * 16;
  const int n0 = blockIdx.y * 64;
  v8f acc[4] = {};
  gemm_128x64_core(xn + (size_t)m0 * 1024, wqkv + (size_t)n0 * 1024,
                   &wtile[0][0], acc, lane, threadIdx.x);

  const int region = n0 >> 10;        // 0=q,1=k,2=v
  const int nr = n0 & 1023;
  const int hH = nr >> 6;
  const int b = m0 >> 10;
  const int tok0 = m0 & 1023;
  const int hi = (lane >> 4) & 1;
  const float* bias = (region == 0) ? bq : (region == 1) ? bk : bv;
#pragma unroll
  for (int t = 0; t < 4; ++t) {
    int dk = t * 16 + (lane & 15);
    float bsc = bias[nr + dk];
    if (region < 2) {
      unsigned short* dst = (region == 0) ? q_out : k_out;
#pragma unroll
      for (int r = 0; r < 8; ++r) {
        float vv = acc[t][r] + bsc;
        dst[((size_t)(b * 16 + hH) * 1024 + tok0 + r + 8 * hi) * 64 + dk] = f2bf(vv);
      }
    } else {
      union { u16x8 v; unsigned short u[8]; } st;
#pragma unroll
      for (int r = 0; r < 8; ++r) st.u[r] = f2bf(acc[t][r] + bsc);
      unsigned short* dst =
          vt_out + ((size_t)(b * 16 + hH) * 64 + dk) * 1024 + tok0 + 8 * hi;
      *(u16x8*)dst = st.v;
    }
  }
}

// ---------------------------------------------------------------------------
// 4) Flash attention (transposed): S^T = K*Q^T, O^T = V^T * P, ALiBi + causal
//    one wave = 16 query rows of one (b,h).
// ---------------------------------------------------------------------------
__global__ __launch_bounds__(256) void attn_kernel(
    const unsigned short* __restrict__ qb,   // [B*H, S, 64]
    const unsigned short* __restrict__ kb,   // [B*H, S, 64]
    const unsigned short* __restrict__ vtb,  // [B*H, 64, S]
    unsigned short* __restrict__ ob) {       // [B, S, 1024] bf16
  const int lane = threadIdx.x & 31;
  const int wave = threadIdx.x >> 5;
  const int bh = blockIdx.x;
  const int b = bh >> 4, h = bh & 15;
  const int q0 = (blockIdx.y * 8 + wave) * 16;
  const float slope = exp2f(-0.5f * (float)(h + 1));
  const float scale = 0.125f;  // 1/sqrt(64)
  const int hi = (lane >> 4) & 1;
  const int qcol = lane & 15;
  const int irow = q0 + qcol;

  const unsigned short* qp = qb + ((size_t)bh * 1024 + q0) * 64;
  const unsigned short* kp = kb + (size_t)bh * 1024 * 64;
  const unsigned short* vp = vtb + (size_t)bh * 64 * 1024;

  v16bf qf0 = load_frag_rm(qp, 64, lane);
  v16bf qf1 = load_frag_rm(qp + 32, 64, lane);

  v8f a0 = {}, a1 = {}, a2 = {}, a3 = {};
  float m = -3.0e38f, l = 0.0f;

  const int nkb = (q0 + 16 + 31) >> 5;  // 32-key blocks, causal cutoff
  for (int t = 0; t < nkb; ++t) {
    const int key0 = t * 32;
    if (t + 1 < nkb) {  // prefetch next K/V block (global_prefetch_b8)
      __builtin_prefetch(kp + (size_t)(key0 + 32) * 64 + (lane << 2), 0, 0);
      __builtin_prefetch(vp + (size_t)(lane & 3) * 16 * 1024 + key0 + 32, 0, 0);
    }
    v8f s0 = {}, s1 = {};
    {
      v16bf ka0 = load_frag_rm(kp + (size_t)key0 * 64, 64, lane);
      v16bf ka1 = load_frag_rm(kp + (size_t)key0 * 64 + 32, 64, lane);
      s0 = wmma_bf16(ka0, qf0, s0);
      s0 = wmma_bf16(ka1, qf1, s0);
      v16bf kb0 = load_frag_rm(kp + (size_t)(key0 + 16) * 64, 64, lane);
      v16bf kb1 = load_frag_rm(kp + (size_t)(key0 + 16) * 64 + 32, 64, lane);
      s1 = wmma_bf16(kb0, qf0, s1);
      s1 = wmma_bf16(kb1, qf1, s1);
    }
    float sv0[8], sv1[8], lmax = -3.0e38f;
#pragma unroll
    for (int r = 0; r < 8; ++r) {
      int j0 = key0 + r + 8 * hi;
      int j1 = j0 + 16;
      float v0 = s0[r] * scale + slope * (float)j0;
      float v1 = s1[r] * scale + slope * (float)j1;
      if (j0 > irow) v0 = NEGBIG;
      if (j1 > irow) v1 = NEGBIG;
      sv0[r] = v0; sv1[r] = v1;
      lmax = fmaxf(lmax, fmaxf(v0, v1));
    }
    lmax = fmaxf(lmax, __shfl_xor(lmax, 16, 32));
    float mnew = fmaxf(m, lmax);
    float corr = __expf(m - mnew);
    float rsum = 0.0f;
    union { v16bf v; unsigned short u[16]; } pf;
#pragma unroll
    for (int r = 0; r < 8; ++r) {
      float p0 = __expf(sv0[r] - mnew);
      float p1 = __expf(sv1[r] - mnew);
      rsum += p0 + p1;
      pf.u[r] = f2bf(p0);       // B-frag elem e   <- tile0 C-reg e (same lane)
      pf.u[8 + r] = f2bf(p1);   // B-frag elem 8+e <- tile1 C-reg e (same lane)
    }
    rsum += __shfl_xor(rsum, 16, 32);
    l = l * corr + rsum;
    m = mnew;
    a0 = a0 * corr; a1 = a1 * corr; a2 = a2 * corr; a3 = a3 * corr;
    a0 = wmma_bf16(load_frag_rm(vp + 0 * 16 * 1024 + key0, 1024, lane), pf.v, a0);
    a1 = wmma_bf16(load_frag_rm(vp + 1 * 16 * 1024 + key0, 1024, lane), pf.v, a1);
    a2 = wmma_bf16(load_frag_rm(vp + 2 * 16 * 1024 + key0, 1024, lane), pf.v, a2);
    a3 = wmma_bf16(load_frag_rm(vp + 3 * 16 * 1024 + key0, 1024, lane), pf.v, a3);
  }
  float inv = (l > 0.f) ? (1.0f / l) : 0.0f;
  a0 = a0 * inv; a1 = a1 * inv; a2 = a2 * inv; a3 = a3 * inv;

  unsigned short* op =
      ob + ((size_t)(b * 1024 + q0 + qcol)) * 1024 + h * 64 + 8 * hi;
  union { u16x8 v; unsigned short u[8]; } st;
#pragma unroll
  for (int r = 0; r < 8; ++r) st.u[r] = f2bf(a0[r]);
  *(u16x8*)(op + 0) = st.v;
#pragma unroll
  for (int r = 0; r < 8; ++r) st.u[r] = f2bf(a1[r]);
  *(u16x8*)(op + 16) = st.v;
#pragma unroll
  for (int r = 0; r < 8; ++r) st.u[r] = f2bf(a2[r]);
  *(u16x8*)(op + 32) = st.v;
#pragma unroll
  for (int r = 0; r < 8; ++r) st.u[r] = f2bf(a3[r]);
  *(u16x8*)(op + 48) = st.v;
}

// ---------------------------------------------------------------------------
// 5) out-proj GEMM + residual:  y = x + o_attn * wo^T + bo   (fp32 out)
// ---------------------------------------------------------------------------
__global__ __launch_bounds__(256) void out_gemm_kernel(
    const unsigned short* __restrict__ oat, const unsigned short* __restrict__ wo,
    const float* __restrict__ bo, const float* __restrict__ x,
    float* __restrict__ y) {
  __shared__ unsigned short wtile[2][64 * 128];
  const int lane = threadIdx.x & 31, wave = threadIdx.x >> 5;
  const int m0 = blockIdx.x * 128 + wave * 16;
  const int n0 = blockIdx.y * 64;
  v8f acc[4] = {};
  gemm_128x64_core(oat + (size_t)m0 * 1024, wo + (size_t)n0 * 1024,
                   &wtile[0][0], acc, lane, threadIdx.x);

  const int hi = (lane >> 4) & 1;
#pragma unroll
  for (int t = 0; t < 4; ++t) {
    int n = n0 + t * 16 + (lane & 15);
    float bsc = bo[n];
#pragma unroll
    for (int r = 0; r < 8; ++r) {
      int mr = m0 + r + 8 * hi;
      y[(size_t)mr * 1024 + n] = acc[t][r] + bsc + x[(size_t)mr * 1024 + n];
    }
  }
}

// ---------------------------------------------------------------------------
// 6) decode + transpose: out[b][f][s] = y[b][s][:] . dec_w[f][:] + dec_b[f]
// ---------------------------------------------------------------------------
__global__ __launch_bounds__(256) void dec_kernel(
    const float* __restrict__ y, const float* __restrict__ dw,
    const float* __restrict__ db, float* __restrict__ out) {
  int tok = blockIdx.x;
  int b = tok >> 10, sidx = tok & 1023;
  __shared__ float ysh[1024];
  __shared__ float part[256];
  for (int i = threadIdx.x; i < 1024; i += 256)
    ysh[i] = y[(size_t)tok * 1024 + i];
  __syncthreads();
  int f = threadIdx.x & 15, seg = threadIdx.x >> 4;
  const float* wr = dw + f * 1024 + seg * 64;
  float s = 0.f;
#pragma unroll
  for (int i = 0; i < 64; ++i) s += ysh[seg * 64 + i] * wr[i];
  part[threadIdx.x] = s;
  __syncthreads();
  if (threadIdx.x < 16) {
    float acc = db[threadIdx.x];
    for (int sg = 0; sg < 16; ++sg) acc += part[sg * 16 + threadIdx.x];
    out[((size_t)b * 16 + threadIdx.x) * 1024 + sidx] = acc;
  }
}

// ---------------------------------------------------------------------------
// Launch
// ---------------------------------------------------------------------------
extern "C" void kernel_launch(void* const* d_in, const int* in_sizes, int n_in,
                              void* d_out, int out_size, void* d_ws, size_t ws_size,
                              hipStream_t stream) {
  const float* z       = (const float*)d_in[0];
  const float* r_traj  = (const float*)d_in[1];
  const float* dense_w = (const float*)d_in[2];
  const float* dense_b = (const float*)d_in[3];
  const float* ln_g    = (const float*)d_in[4];
  const float* ln_b    = (const float*)d_in[5];
  const float* wq      = (const float*)d_in[6];
  const float* bq      = (const float*)d_in[7];
  const float* wk      = (const float*)d_in[8];
  const float* bk      = (const float*)d_in[9];
  const float* wv      = (const float*)d_in[10];
  const float* bv      = (const float*)d_in[11];
  const float* wo      = (const float*)d_in[12];
  const float* bo      = (const float*)d_in[13];
  const float* dec_w   = (const float*)d_in[14];
  const float* dec_b   = (const float*)d_in[15];

  char* ws = (char*)d_ws;
  auto alloc = [&](size_t bytes) {
    char* p = ws;
    ws += (bytes + 255) & ~(size_t)255;
    return p;
  };
  const size_t NT = 8192;  // B*S tokens
  float*          zp    = (float*)alloc(8 * 1024 * 4);
  float*          x     = (float*)alloc(NT * 1024 * 4);
  unsigned short* xn    = (unsigned short*)alloc(NT * 1024 * 2);
  unsigned short* wqkvb = (unsigned short*)alloc((size_t)3072 * 1024 * 2);
  unsigned short* wob   = (unsigned short*)alloc((size_t)1024 * 1024 * 2);
  unsigned short* qbuf  = (unsigned short*)alloc(NT * 1024 * 2);
  unsigned short* kbuf  = (unsigned short*)alloc(NT * 1024 * 2);
  unsigned short* vtbuf = (unsigned short*)alloc(NT * 1024 * 2);
  unsigned short* oat   = (unsigned short*)alloc(NT * 1024 * 2);
  float*          y     = (float*)alloc(NT * 1024 * 4);

  const int NW = 1024 * 1024;
  cvt_bf16_kernel<<<(NW + 255) / 256, 256, 0, stream>>>(wq, wqkvb, NW);
  cvt_bf16_kernel<<<(NW + 255) / 256, 256, 0, stream>>>(wk, wqkvb + NW, NW);
  cvt_bf16_kernel<<<(NW + 255) / 256, 256, 0, stream>>>(wv, wqkvb + 2 * NW, NW);
  cvt_bf16_kernel<<<(NW + 255) / 256, 256, 0, stream>>>(wo, wob, NW);

  zpart_kernel<<<dim3(4, 8), 256, 0, stream>>>(z, dense_w, dense_b, zp);
  dense_ln_kernel<<<8192, 256, 0, stream>>>(r_traj, dense_w, zp, ln_g, ln_b, x, xn);
  qkv_gemm_kernel<<<dim3(64, 48), 256, 0, stream>>>(xn, wqkvb, bq, bk, bv,
                                                    qbuf, kbuf, vtbuf);
  attn_kernel<<<dim3(128, 8), 256, 0, stream>>>(qbuf, kbuf, vtbuf, oat);
  out_gemm_kernel<<<dim3(64, 16), 256, 0, stream>>>(oat, wob, bo, x, y);
  dec_kernel<<<8192, 256, 0, stream>>>(y, dec_w, dec_b, (float*)d_out);
}
